// MultiHeadAttention_26070451487284
// MI455X (gfx1250) — matrix operand order repaired and checked by reading, not compiled, not run on hardware
//
#include <hip/hip_runtime.h>
#include <hip/hip_bf16.h>

typedef __attribute__((ext_vector_type(16))) __bf16 v16bf;
typedef __attribute__((ext_vector_type(8)))  float  v8f;
typedef __attribute__((ext_vector_type(4)))  unsigned int v4u;
typedef __attribute__((ext_vector_type(8)))  int    v8i_t;
typedef __attribute__((ext_vector_type(4)))  int    v4i_t;

struct U8x32  { uint4 a, b; };        // 32 bytes == one v16bf fragment
struct U32x8b { unsigned u[8]; };     // 32 bytes, for reg-assembled fragments

// ---- f32 -> bf16 (packed pair), hardware cvt if available -----------------
#if __has_builtin(__builtin_amdgcn_cvt_pk_bf16_f32)
typedef __attribute__((ext_vector_type(2))) __bf16 v2bf;
static __device__ __forceinline__ unsigned pack2(float x, float y) {
    v2bf r = __builtin_amdgcn_cvt_pk_bf16_f32(x, y);
    return __builtin_bit_cast(unsigned, r);
}
#else
static __device__ __forceinline__ unsigned short f2bf_(float f) {
    unsigned u = __builtin_bit_cast(unsigned, f);
    u += 0x7FFFu + ((u >> 16) & 1u);
    return (unsigned short)(u >> 16);
}
static __device__ __forceinline__ unsigned pack2(float x, float y) {
    return (unsigned)f2bf_(x) | ((unsigned)f2bf_(y) << 16);
}
#endif
static __device__ __forceinline__ unsigned short f2bf1(float v) {
    return (unsigned short)(pack2(v, 0.f) & 0xFFFFu);
}
static __device__ __forceinline__ v16bf make_frag(const unsigned short* p0,
                                                  const unsigned short* p1) {
    U8x32 u;
    u.a = *(const uint4*)p0;
    u.b = *(const uint4*)p1;
    return __builtin_bit_cast(v16bf, u);
}

// ---------------------------------------------------------------------------
// bulk f32 -> bf16 conversion (8 elements / thread)
// ---------------------------------------------------------------------------
__global__ __launch_bounds__(256) void cvt_bf16(const float* __restrict__ src,
                                                unsigned short* __restrict__ dst,
                                                int n8) {
    int i = blockIdx.x * 256 + threadIdx.x;
    if (i >= n8) return;
    const float4* s = (const float4*)src;
    float4 a = s[2 * i], b = s[2 * i + 1];
    uint4 o;
    o.x = pack2(a.x, a.y); o.y = pack2(a.z, a.w);
    o.z = pack2(b.x, b.y); o.w = pack2(b.z, b.w);
    ((uint4*)dst)[i] = o;
}

// ---------------------------------------------------------------------------
// Tensor Data Mover: generic 2D bf16 tile load with LDS row padding.
// pad_code c: pad after 2^(c+1) DWORDs; pad_amt code a: (a+1) DWORDs.
// ---------------------------------------------------------------------------
#define ASTR 40   // GEMM: 32 + 8 halves
#define KSTR 72   // attention: 64 + 8 halves
#define USE_TDM (__has_builtin(__builtin_amdgcn_tensor_load_to_lds) && \
                 __has_builtin(__builtin_amdgcn_s_wait_tensorcnt))

#if USE_TDM
static __device__ __forceinline__ void tdm_load_2d(unsigned lds_addr,
                                                   unsigned long long gaddr,
                                                   int tensor_d0, int tensor_d1,
                                                   int tile_d0, int tile_d1,
                                                   int stride0,
                                                   int pad_code, int pad_amt) {
    v4u g0;
    g0[0] = 1u;                                    // count=1, user mode
    g0[1] = lds_addr;                              // LDS byte address
    g0[2] = (unsigned)(gaddr & 0xFFFFFFFFull);     // global_addr lo
    g0[3] = (unsigned)((gaddr >> 32) & 0x1FFFFFFull) | 0x80000000u; // hi | type=2
    v8i_t g1;
    g1[0] = (1 << 16)            // data_size = 2 bytes
          | (1 << 20)            // pad_enable
          | (pad_code << 22)
          | (pad_amt << 25);
    g1[1] = (tensor_d0 & 0xFFFF) << 16;
    g1[2] = ((tensor_d0 >> 16) & 0xFFFF) | ((tensor_d1 & 0xFFFF) << 16);
    g1[3] = ((tensor_d1 >> 16) & 0xFFFF) | ((tile_d0 & 0xFFFF) << 16);
    g1[4] = tile_d1 & 0xFFFF;    // tile_dim2 = 0
    g1[5] = stride0;
    g1[6] = 0;
    g1[7] = 0;
    v4i_t gz = {0, 0, 0, 0};
#if defined(__clang_major__) && __clang_major__ >= 23
    v8i_t gz8 = {0, 0, 0, 0, 0, 0, 0, 0};
    __builtin_amdgcn_tensor_load_to_lds(g0, g1, gz, gz, gz8, 0);
#else
    __builtin_amdgcn_tensor_load_to_lds(g0, g1, gz, gz, 0);
#endif
}
#endif

// ---------------------------------------------------------------------------
// GEMM: C[8192,1024] = A[8192,1024](bf16) x W^T (W[1024,1024] bf16 [out,in])
//       + bias(f32), epilogue scatter by mode.
// mode 0: Q -> bf16 [B,H,S,Dk] scaled by log2(e)/8   mode 1: K -> [B,H,S,Dk]
// mode 2: V -> bf16 [B,H,Dk,S] (transposed)          mode 3: f32 [8192,1024]
// ---------------------------------------------------------------------------
__global__ __launch_bounds__(256) void gemm_bf16(
    const unsigned short* __restrict__ A, const unsigned short* __restrict__ W,
    const float* __restrict__ bias, void* __restrict__ outp, int mode)
{
    __shared__ alignas(16) unsigned short At[2][128 * ASTR];
    __shared__ alignas(16) unsigned short Bt[2][128 * ASTR];

    const int t    = threadIdx.x;
    const int lane = t & 31;
    const int wave = t >> 5;
    const int l15  = lane & 15;
    const int hi   = lane >> 4;
    const int m0   = blockIdx.x * 128;
    const int n0   = blockIdx.y * 128;
    const int waveM = (wave >> 1) * 32;
    const int waveN = (wave & 1) * 64;

    v8f c[2][4];
#pragma unroll
    for (int ms = 0; ms < 2; ++ms)
#pragma unroll
        for (int ns = 0; ns < 4; ++ns)
#pragma unroll
            for (int i = 0; i < 8; ++i) c[ms][ns][i] = 0.f;

#if USE_TDM
    if (wave == 0) {
        tdm_load_2d((unsigned)(size_t)&At[0][0],
                    (unsigned long long)(size_t)(A + (size_t)m0 * 1024),
                    1024, 8192, 32, 128, 1024, 3, 3);
        tdm_load_2d((unsigned)(size_t)&Bt[0][0],
                    (unsigned long long)(size_t)(W + (size_t)n0 * 1024),
                    1024, 1024, 32, 128, 1024, 3, 3);
    }
#endif

    for (int k0 = 0; k0 < 1024; k0 += 32) {
        const int pb = (k0 >> 5) & 1;
#if USE_TDM
        if (wave == 0) __builtin_amdgcn_s_wait_tensorcnt(0);
        __syncthreads();
        if (wave == 0 && k0 + 32 < 1024) {
            tdm_load_2d((unsigned)(size_t)&At[pb ^ 1][0],
                        (unsigned long long)(size_t)(A + (size_t)m0 * 1024 + (k0 + 32)),
                        1024, 8192, 32, 128, 1024, 3, 3);
            tdm_load_2d((unsigned)(size_t)&Bt[pb ^ 1][0],
                        (unsigned long long)(size_t)(W + (size_t)n0 * 1024 + (k0 + 32)),
                        1024, 1024, 32, 128, 1024, 3, 3);
        }
#else
        __syncthreads();
        {
            int row = t >> 1;
            int cb  = (t & 1) * 16;
            const size_t ga = (size_t)(m0 + row) * 1024 + k0 + cb;
            const size_t gw = (size_t)(n0 + row) * 1024 + k0 + cb;
            *(uint4*)&At[pb][row * ASTR + cb]     = *(const uint4*)&A[ga];
            *(uint4*)&At[pb][row * ASTR + cb + 8] = *(const uint4*)&A[ga + 8];
            *(uint4*)&Bt[pb][row * ASTR + cb]     = *(const uint4*)&W[gw];
            *(uint4*)&Bt[pb][row * ASTR + cb + 8] = *(const uint4*)&W[gw + 8];
        }
        __syncthreads();
#endif
        v16bf af[2];
#pragma unroll
        for (int ms = 0; ms < 2; ++ms) {
            int row  = waveM + ms * 16 + l15;
            int koff = hi * 8;
            af[ms] = make_frag(&At[pb][row * ASTR + koff],
                               &At[pb][row * ASTR + koff + 16]);
        }
        v16bf bfr[4];
#pragma unroll
        for (int ns = 0; ns < 4; ++ns) {
            int col  = waveN + ns * 16 + l15;
            int koff = hi * 8;
            bfr[ns] = make_frag(&Bt[pb][col * ASTR + koff],
                                &Bt[pb][col * ASTR + koff + 16]);
        }
#pragma unroll
        for (int ms = 0; ms < 2; ++ms)
#pragma unroll
            for (int ns = 0; ns < 4; ++ns)
                c[ms][ns] = __builtin_amdgcn_wmma_f32_16x16x32_bf16(
                    false, af[ms], false, bfr[ns], (short)0, c[ms][ns], false, false);
    }

#pragma unroll
    for (int ns = 0; ns < 4; ++ns) {
        int colg = n0 + waveN + ns * 16 + l15;
        float bv = bias[colg];
#pragma unroll
        for (int ms = 0; ms < 2; ++ms) {
            int rowg0 = m0 + waveM + ms * 16 + hi * 8;
#pragma unroll
            for (int r = 0; r < 8; ++r) {
                float v  = c[ms][ns][r] + bv;
                int rowg = rowg0 + r;
                if (mode == 3) {
                    ((float*)outp)[(size_t)rowg * 1024 + colg] = v;
                } else {
                    if (mode == 0) v *= 0.18033688f;   // log2(e)/sqrt(64)
                    int b = rowg >> 11, s = rowg & 2047;
                    int h = colg >> 6,  d = colg & 63;
                    unsigned short o16 = f2bf1(v);
                    if (mode == 2)
                        ((unsigned short*)outp)[(((size_t)(b * 16 + h) * 64 + d) * 2048) + s] = o16;
                    else
                        ((unsigned short*)outp)[(((size_t)(b * 16 + h) * 2048 + s) * 64) + d] = o16;
                }
            }
        }
    }
}

// ---------------------------------------------------------------------------
// Flash attention, transposed formulation:
//   S^T = K . Q^T   (M=key, N=query)  -> per-lane scalar softmax state
//   ctx^T = V^T . P^T                 -> contiguous b128 ctx stores
// K/V tiles (64x64) staged per-workgroup via TDM, double buffered.
// ---------------------------------------------------------------------------
__global__ __launch_bounds__(128) void attn_kernel(
    const unsigned short* __restrict__ Qb, const unsigned short* __restrict__ Kb,
    const unsigned short* __restrict__ Vt, unsigned short* __restrict__ ctx)
{
    __shared__ alignas(16) unsigned short Kt[2][64 * KSTR];
    __shared__ alignas(16) unsigned short Vl[2][64 * KSTR];

    const int t    = threadIdx.x;
    const int lane = t & 31;
    const int w    = t >> 5;
    const int l15  = lane & 15;
    const int hi   = lane >> 4;
    const int qt   = blockIdx.x;      // 0..31
    const int bh   = blockIdx.y;      // 0..63
    const int b    = bh >> 4, h = bh & 15;
    const size_t baseQK = (size_t)bh * 2048 * 64;
    const size_t baseV  = (size_t)bh * 64 * 2048;

    // Q^T B-fragments (lane = query column), kept in registers
    const int qrow = qt * 64 + w * 16 + l15;
    v16bf bq[2];
#pragma unroll
    for (int f = 0; f < 2; ++f) {
        const unsigned short* p = &Qb[baseQK + (size_t)qrow * 64 + f * 32 + hi * 8];
        bq[f] = make_frag(p, p + 16);
    }

    float mrun = -3.0e38f, lrun = 0.f;
    v8f acc[4];
#pragma unroll
    for (int j = 0; j < 4; ++j)
#pragma unroll
        for (int r = 0; r < 8; ++r) acc[j][r] = 0.f;

#if USE_TDM
    if (w == 0) {
        tdm_load_2d((unsigned)(size_t)&Kt[0][0],
                    (unsigned long long)(size_t)(Kb + baseQK),
                    64, 2048, 64, 64, 64, 4, 3);
        tdm_load_2d((unsigned)(size_t)&Vl[0][0],
                    (unsigned long long)(size_t)(Vt + baseV),
                    2048, 64, 64, 64, 2048, 4, 3);
    }
#endif

    for (int kb = 0; kb < 2048; kb += 64) {
        const int pb = (kb >> 6) & 1;
#if USE_TDM
        if (w == 0) __builtin_amdgcn_s_wait_tensorcnt(0);
        __syncthreads();
        if (w == 0 && kb + 64 < 2048) {
            tdm_load_2d((unsigned)(size_t)&Kt[pb ^ 1][0],
                        (unsigned long long)(size_t)(Kb + baseQK + (size_t)(kb + 64) * 64),
                        64, 2048, 64, 64, 64, 4, 3);
            tdm_load_2d((unsigned)(size_t)&Vl[pb ^ 1][0],
                        (unsigned long long)(size_t)(Vt + baseV + (size_t)(kb + 64)),
                        2048, 64, 64, 64, 2048, 4, 3);
        }
#else
        __syncthreads();
        {
            int row = t >> 1;
            int cb  = (t & 1) * 32;
            *(uint4*)&Kt[pb][row * KSTR + cb]     = *(const uint4*)&Kb[baseQK + (size_t)(kb + row) * 64 + cb];
            *(uint4*)&Kt[pb][row * KSTR + cb + 8] = *(const uint4*)&Kb[baseQK + (size_t)(kb + row) * 64 + cb + 8];
            *(uint4*)&Vl[pb][row * KSTR + cb]     = *(const uint4*)&Vt[baseV + (size_t)row * 2048 + kb + cb];
            *(uint4*)&Vl[pb][row * KSTR + cb + 8] = *(const uint4*)&Vt[baseV + (size_t)row * 2048 + kb + cb + 8];
        }
        __syncthreads();
#endif
        // S^T fragments: sc[n] covers keys kb+n*16.. (M), queries (N=lane)
        v8f sc[4];
#pragma unroll
        for (int n = 0; n < 4; ++n) {
            v8f z;
#pragma unroll
            for (int i = 0; i < 8; ++i) z[i] = 0.f;
#pragma unroll
            for (int f = 0; f < 2; ++f) {
                const unsigned short* p = &Kt[pb][(n * 16 + l15) * KSTR + f * 32 + hi * 8];
                v16bf ak = make_frag(p, p + 16);
                z = __builtin_amdgcn_wmma_f32_16x16x32_bf16(
                        false, ak, false, bq[f], (short)0, z, false, false);
            }
            sc[n] = z;
        }

        // per-lane softmax (query on lane axis); keys split with partner lane^16
        float mx = -3.0e38f;
#pragma unroll
        for (int n = 0; n < 4; ++n)
#pragma unroll
            for (int r = 0; r < 8; ++r) mx = fmaxf(mx, sc[n][r]);
        mx = fmaxf(mx, __shfl_xor(mx, 16, 32));
        float mn = fmaxf(mrun, mx);
        float al = __builtin_exp2f(mrun - mn);
        mrun = mn;
        float p4[4][8];
        float sum = 0.f;
#pragma unroll
        for (int n = 0; n < 4; ++n)
#pragma unroll
            for (int r = 0; r < 8; ++r) {
                p4[n][r] = __builtin_exp2f(sc[n][r] - mn);
                sum += p4[n][r];
            }
        sum += __shfl_xor(sum, 16, 32);
        lrun = lrun * al + sum;
#pragma unroll
        for (int j = 0; j < 4; ++j)
#pragma unroll
            for (int r = 0; r < 8; ++r) acc[j][r] *= al;

        // P^T -> B-layout fragments via packed lane^16 exchange (no LDS)
        unsigned pk[4][4];
#pragma unroll
        for (int n = 0; n < 4; ++n)
#pragma unroll
            for (int i = 0; i < 4; ++i)
                pk[n][i] = pack2(p4[n][2 * i], p4[n][2 * i + 1]);

#pragma unroll
        for (int g = 0; g < 2; ++g) {   // key 32-blocks: fragments (2g, 2g+1)
            U32x8b bw;
#pragma unroll
            for (int i = 0; i < 4; ++i) {
                unsigned qa = __shfl_xor(pk[2 * g][i],     16, 32);
                unsigned qb = __shfl_xor(pk[2 * g + 1][i], 16, 32);
                bw.u[i]     = hi ? qb                : pk[2 * g][i];
                bw.u[4 + i] = hi ? pk[2 * g + 1][i]  : qa;
            }
            v16bf bp = __builtin_bit_cast(v16bf, bw);
#pragma unroll
            for (int j = 0; j < 4; ++j) {
                const unsigned short* p = &Vl[pb][(j * 16 + l15) * KSTR + g * 32 + hi * 16];
                v16bf av = make_frag(p, p + 8);
                acc[j] = __builtin_amdgcn_wmma_f32_16x16x32_bf16(
                            false, av, false, bp, (short)0, acc[j], false, false);
            }
        }
    }

    // normalize; ctx^T C-layout -> contiguous bf16 stores (lane = query)
    const float rinv = 1.0f / lrun;
    const size_t crow = ((size_t)(b * 2048 + qt * 64 + w * 16 + l15)) * 1024 + h * 64;
#pragma unroll
    for (int j = 0; j < 4; ++j) {
        uint4 o;
        o.x = pack2(acc[j][0] * rinv, acc[j][1] * rinv);
        o.y = pack2(acc[j][2] * rinv, acc[j][3] * rinv);
        o.z = pack2(acc[j][4] * rinv, acc[j][5] * rinv);
        o.w = pack2(acc[j][6] * rinv, acc[j][7] * rinv);
        *(uint4*)&ctx[crow + j * 16 + hi * 8] = o;
    }
}

// ---------------------------------------------------------------------------
extern "C" void kernel_launch(void* const* d_in, const int* in_sizes, int n_in,
                              void* d_out, int out_size, void* d_ws, size_t ws_size,
                              hipStream_t stream) {
    const float* x  = (const float*)d_in[0];
    const float* Wq = (const float*)d_in[1];
    const float* bq = (const float*)d_in[2];
    const float* Wk = (const float*)d_in[3];
    const float* bk = (const float*)d_in[4];
    const float* Wv = (const float*)d_in[5];
    const float* bv = (const float*)d_in[6];
    const float* Wo = (const float*)d_in[7];
    const float* bo = (const float*)d_in[8];
    float* out = (float*)d_out;

    char* ws = (char*)d_ws;
    unsigned short* xb  = (unsigned short*)(ws);                      // 16 MB
    unsigned short* Wqb = (unsigned short*)(ws + (size_t)16777216);   //  2 MB
    unsigned short* Wkb = (unsigned short*)(ws + (size_t)18874368);
    unsigned short* Wvb = (unsigned short*)(ws + (size_t)20971520);
    unsigned short* Wob = (unsigned short*)(ws + (size_t)23068672);
    unsigned short* Qb  = (unsigned short*)(ws + (size_t)25165824);   // 16 MB
    unsigned short* Kb  = (unsigned short*)(ws + (size_t)41943040);   // 16 MB
    unsigned short* Vt  = (unsigned short*)(ws + (size_t)58720256);   // 16 MB
    unsigned short* cxb = (unsigned short*)(ws + (size_t)75497472);   // 16 MB

    cvt_bf16<<<4096, 256, 0, stream>>>(x,  xb,  1048576);
    cvt_bf16<<<512,  256, 0, stream>>>(Wq, Wqb, 131072);
    cvt_bf16<<<512,  256, 0, stream>>>(Wk, Wkb, 131072);
    cvt_bf16<<<512,  256, 0, stream>>>(Wv, Wvb, 131072);
    cvt_bf16<<<512,  256, 0, stream>>>(Wo, Wob, 131072);

    dim3 gg(64, 8, 1), gb(256);
    gemm_bf16<<<gg, gb, 0, stream>>>(xb, Wqb, bq, (void*)Qb, 0);
    gemm_bf16<<<gg, gb, 0, stream>>>(xb, Wkb, bk, (void*)Kb, 1);
    gemm_bf16<<<gg, gb, 0, stream>>>(xb, Wvb, bv, (void*)Vt, 2);
    attn_kernel<<<dim3(32, 64), 128, 0, stream>>>(Qb, Kb, Vt, cxb);
    gemm_bf16<<<gg, gb, 0, stream>>>(cxb, Wob, bo, (void*)out, 3);
}
